// WeightedSelfAttention_62697932587414
// MI455X (gfx1250) — matrix operand order repaired and checked by reading, not compiled, not run on hardware
//
#include <hip/hip_runtime.h>

typedef __attribute__((ext_vector_type(16))) _Float16 v16h;
typedef __attribute__((ext_vector_type(8)))  _Float16 v8h;
typedef __attribute__((ext_vector_type(8)))  float    v8f;
typedef __attribute__((ext_vector_type(4)))  float    v4f;

static constexpr int NSEQ = 900;
static constexpr int NPAD = 912;    // padded key stride for global v^T rows
static constexpr int NB   = 16;
static constexpr int CH   = 256;
static constexpr int NH   = 8;
static constexpr int HD   = 32;
// LDS staging strides (halves), chosen for conflict-free 16B ds loads
static constexpr int KLS  = 40;     // K row stride: 80B -> bank step 20, all 16 lanes distinct
static constexpr int VLS  = 920;    // V^T row stride: 1840B -> bank step 12, all 16 lanes distinct
static constexpr int KL_HALVES = NSEQ * KLS;            // 36000
static constexpr int VL_HALVES = HD * VLS;              // 29440
static constexpr int LDS_HALVES = KL_HALVES + VL_HALVES + 128;  // +slack for tail over-read
#define ATT_SCALE 0.17677669529663687f  // 32^-0.5

// ---------------------------------------------------------------------------
// Kernel 1: fused Q/K/V projections.  out = X @ W^T + b
// q (pre-scaled), k -> [B,H,N,D] f16 ; v -> [B,H,D,NPAD] f16 (transposed).
// Each 16x16 tile covers ONE sequence position (n = mtile), all 16 batches.
// ---------------------------------------------------------------------------
__global__ __launch_bounds__(128) void qkv_proj_kernel(
    const float* __restrict__ qin, const float* __restrict__ kin, const float* __restrict__ vin,
    const float* __restrict__ Wq, const float* __restrict__ bq,
    const float* __restrict__ Wk, const float* __restrict__ bk,
    const float* __restrict__ Wv, const float* __restrict__ bv,
    _Float16* __restrict__ q16, _Float16* __restrict__ k16, _Float16* __restrict__ vT16)
{
  const int mat = blockIdx.z;
  const float* X    = (mat == 0) ? qin : (mat == 1) ? kin : vin;
  const float* W    = (mat == 0) ? Wq  : (mat == 1) ? Wk  : Wv;
  const float* bias = (mat == 0) ? bq  : (mat == 1) ? bk  : bv;

  const int lane = threadIdx.x;        // 0..31
  const int h  = lane >> 4;            // half-wave
  const int ln = lane & 15;
  const int ntile = blockIdx.x * 4 + threadIdx.y;  // 0..15
  const int mtile = blockIdx.y;                    // 0..899
  const int rowbase = mtile * 16;
  const int colbase = ntile * 16;

  v8f acc = {};
  const float* arowb = X + (size_t)(rowbase + ln) * CH;
  const float* browb = W + (size_t)(colbase + ln) * CH;
  for (int kc = 0; kc < CH; kc += 32) {
    v16h a, bfr;
    // A (16x32 f16): lane(h,m=ln), elem j -> K = (j<8 ? j : j+8) + 8h
    v4f a0 = *(const v4f*)(arowb + kc + h * 8);
    v4f a1 = *(const v4f*)(arowb + kc + h * 8 + 4);
    v4f a2 = *(const v4f*)(arowb + kc + h * 8 + 16);
    v4f a3 = *(const v4f*)(arowb + kc + h * 8 + 20);
    #pragma unroll
    for (int j = 0; j < 4; ++j) {
      a[j]      = (_Float16)a0[j];
      a[j + 4]  = (_Float16)a1[j];
      a[j + 8]  = (_Float16)a2[j];
      a[j + 12] = (_Float16)a3[j];
    }
    // B (32x16 f16): lane(h,n=ln), elem j -> K = (j<8 ? j : j+8) + 8h
    v4f b0 = *(const v4f*)(browb + kc + h * 8);
    v4f b1 = *(const v4f*)(browb + kc + h * 8 + 4);
    v4f b2 = *(const v4f*)(browb + kc + h * 8 + 16);
    v4f b3 = *(const v4f*)(browb + kc + h * 8 + 20);
    #pragma unroll
    for (int j = 0; j < 4; ++j) {
      bfr[j]      = (_Float16)b0[j];
      bfr[j + 4]  = (_Float16)b1[j];
      bfr[j + 8]  = (_Float16)b2[j];
      bfr[j + 12] = (_Float16)b3[j];
    }
    acc = __builtin_amdgcn_wmma_f32_16x16x32_f16(false, a, false, bfr, (short)0, acc, false, false);
  }

  const int o  = colbase + ln;       // output channel
  const int hh = o >> 5;             // head
  const int d  = o & 31;             // dim within head
  const float bval = bias[o];
  // C layout: reg r -> batch b = r + 8h (n == mtile for the whole tile)
  if (mat == 2) {
    unsigned idx = (unsigned)((8 * h) * NH + hh) * (HD * NPAD) + (unsigned)d * NPAD + mtile;
    #pragma unroll
    for (int r = 0; r < 8; ++r) {
      vT16[idx] = (_Float16)(acc[r] + bval);
      idx += NH * HD * NPAD;                       // next batch
    }
  } else {
    _Float16* dst = (mat == 0) ? q16 : k16;
    const float scale = (mat == 0) ? ATT_SCALE : 1.0f;
    unsigned idx = ((unsigned)((8 * h) * NH + hh) * NSEQ + mtile) * HD + d;
    #pragma unroll
    for (int r = 0; r < 8; ++r) {
      dst[idx] = (_Float16)((acc[r] + bval) * scale);
      idx += NH * NSEQ * HD;                       // next batch
    }
  }
}

// ---------------------------------------------------------------------------
// Kernel 2: flash attention, transposed.  S^T = K Q^T, *= prior, online
// softmax, O^T += V^T P^T.  A block stages K and V^T of one (b,h) in LDS
// once; its 8 waves then sweep query tiles reading fragments from LDS.
// 28 full 32-key chunks (no masking) + 1 guarded tail chunk.
// ---------------------------------------------------------------------------
template <bool FULL>
__device__ __forceinline__ void attn_chunk(
    int kc, int h, int ln,
    const _Float16* __restrict__ kl, const _Float16* __restrict__ vl,
    const float* __restrict__ prior_row, const v16h& qfrag,
    v8f& Oa, v8f& Ob, float& m_run, float& l_run)
{
  // ---- K A-fragments (keys x d), two 16-key tiles, from LDS
  v16h ka0, ka1;
  if (FULL) {
    const _Float16* kr0 = kl + (kc + ln) * KLS + h * 8;
    const _Float16* kr1 = kl + (kc + 16 + ln) * KLS + h * 8;
    v8h l0 = *(const v8h*)kr0, h0 = *(const v8h*)(kr0 + 16);
    v8h l1 = *(const v8h*)kr1, h1 = *(const v8h*)(kr1 + 16);
    #pragma unroll
    for (int j = 0; j < 8; ++j) { ka0[j] = l0[j]; ka0[j + 8] = h0[j];
                                  ka1[j] = l1[j]; ka1[j + 8] = h1[j]; }
  } else {
    int key0 = kc + ln;        bool v0 = key0 < NSEQ;
    int key1 = kc + 16 + ln;   bool v1 = key1 < NSEQ;
    const _Float16* kr0 = kl + (v0 ? key0 : 0) * KLS + h * 8;
    const _Float16* kr1 = kl + (v1 ? key1 : 0) * KLS + h * 8;
    _Float16 z = (_Float16)0.0f;
    #pragma unroll
    for (int j = 0; j < 8; ++j) {
      ka0[j]     = v0 ? kr0[j]      : z;
      ka0[j + 8] = v0 ? kr0[16 + j] : z;
      ka1[j]     = v1 ? kr1[j]      : z;
      ka1[j + 8] = v1 ? kr1[16 + j] : z;
    }
  }
  v8f zero = {};
  v8f s0 = __builtin_amdgcn_wmma_f32_16x16x32_f16(false, ka0, false, qfrag, (short)0, zero, false, false);
  v8f s1 = __builtin_amdgcn_wmma_f32_16x16x32_f16(false, ka1, false, qfrag, (short)0, zero, false, false);

  // ---- prior multiply (+ key masking on tail) + chunk max
  float x0[8], x1[8];
  float tmax = -3.0e38f;
  if (FULL) {
    const v4f* pA = (const v4f*)(prior_row + kc + 8 * h);        // keys kc+8h..+7
    const v4f* pB = (const v4f*)(prior_row + kc + 16 + 8 * h);   // keys +16
    v4f pa0 = pA[0], pa1 = pA[1], pb0 = pB[0], pb1 = pB[1];
    #pragma unroll
    for (int r = 0; r < 4; ++r) {
      x0[r]     = s0[r]     * pa0[r];
      x0[r + 4] = s0[r + 4] * pa1[r];
      x1[r]     = s1[r]     * pb0[r];
      x1[r + 4] = s1[r + 4] * pb1[r];
    }
    #pragma unroll
    for (int r = 0; r < 8; ++r) tmax = fmaxf(tmax, fmaxf(x0[r], x1[r]));
  } else {
    #pragma unroll
    for (int r = 0; r < 8; ++r) {
      int keyA = kc + r + 8 * h;
      int keyB = keyA + 16;
      bool vA = keyA < NSEQ;
      bool vB = keyB < NSEQ;
      float pA = vA ? prior_row[keyA] : 0.0f;
      float pB = vB ? prior_row[keyB] : 0.0f;
      x0[r] = vA ? s0[r] * pA : -3.0e38f;
      x1[r] = vB ? s1[r] * pB : -3.0e38f;
      tmax = fmaxf(tmax, fmaxf(x0[r], x1[r]));
    }
  }
  tmax = fmaxf(tmax, __shfl_xor(tmax, 16, 32));   // combine halves of column q

  float mnew  = fmaxf(m_run, tmax);
  float alpha = __expf(m_run - mnew);
  m_run = mnew;

  float p0[8], p1[8];
  float tsum = 0.0f;
  #pragma unroll
  for (int r = 0; r < 8; ++r) {
    p0[r] = __expf(x0[r] - mnew);                 // masked keys -> exp(-huge) = 0
    p1[r] = __expf(x1[r] - mnew);
    tsum += p0[r] + p1[r];
  }
  tsum += __shfl_xor(tsum, 16, 32);
  l_run = l_run * alpha + tsum;

  #pragma unroll
  for (int r = 0; r < 8; ++r) { Oa[r] *= alpha; Ob[r] *= alpha; }

  // ---- P^T B-fragment: lane(h,q), elem j -> local key (j<8 ? j : j+8) + 8h
  v16h pfrag;
  #pragma unroll
  for (int j = 0; j < 8; ++j) {
    pfrag[j]     = (_Float16)p0[j];
    pfrag[j + 8] = (_Float16)p1[j];
  }

  // ---- V^T A-fragments (d x keys), two d tiles, from LDS
  v16h va0, va1;
  const _Float16* vr0 = vl + ln * VLS        + kc + h * 8;
  const _Float16* vr1 = vl + (16 + ln) * VLS + kc + h * 8;
  if (FULL) {
    v8h l0 = *(const v8h*)vr0, h0 = *(const v8h*)(vr0 + 16);
    v8h l1 = *(const v8h*)vr1, h1 = *(const v8h*)(vr1 + 16);
    #pragma unroll
    for (int j = 0; j < 8; ++j) { va0[j] = l0[j]; va0[j + 8] = h0[j];
                                  va1[j] = l1[j]; va1[j + 8] = h1[j]; }
  } else {
    _Float16 z = (_Float16)0.0f;
    #pragma unroll
    for (int j = 0; j < 8; ++j) {
      int keyA = kc + h * 8 + j;
      int keyB = keyA + 16;
      bool vA = keyA < NSEQ;
      bool vB = keyB < NSEQ;
      va0[j]     = vA ? vr0[j]      : z;
      va0[j + 8] = vB ? vr0[16 + j] : z;
      va1[j]     = vA ? vr1[j]      : z;
      va1[j + 8] = vB ? vr1[16 + j] : z;
    }
  }
  Oa = __builtin_amdgcn_wmma_f32_16x16x32_f16(false, va0, false, pfrag, (short)0, Oa, false, false);
  Ob = __builtin_amdgcn_wmma_f32_16x16x32_f16(false, va1, false, pfrag, (short)0, Ob, false, false);
}

__global__ __launch_bounds__(256) void attn_kernel(
    const _Float16* __restrict__ q16, const _Float16* __restrict__ k16,
    const _Float16* __restrict__ vT16, const float* __restrict__ prior,
    _Float16* __restrict__ ctx)
{
  extern __shared__ _Float16 smem[];
  _Float16* kl = smem;                  // K  staged: [900][KLS]
  _Float16* vl = smem + KL_HALVES;      // V^T staged: [32][VLS] (+slack)

  const int bh   = blockIdx.y;          // 0..127
  const int part = blockIdx.x;          // 0..3 : query-tile range
  const int b  = bh >> 3;
  const int hh = bh & 7;
  const int lane = threadIdx.x;
  const int wave = threadIdx.y;         // 0..7
  const int tid  = wave * 32 + lane;
  const int h  = lane >> 4;
  const int ln = lane & 15;

  // ---- cooperative LDS fill: K and V^T of this (b,h), 16B chunks
  {
    const _Float16* kg = k16  + (size_t)bh * NSEQ * HD;
    const _Float16* vg = vT16 + (size_t)bh * HD * NPAD;
    for (int idx = tid; idx < NSEQ * 4; idx += 256) {       // 900 rows x 4 chunks
      int row = idx >> 2, c = (idx & 3) * 8;
      *(v8h*)(kl + row * KLS + c) = *(const v8h*)(kg + row * HD + c);
    }
    for (int idx = tid; idx < HD * 114; idx += 256) {       // 32 rows x 114 chunks
      int row = idx / 114, c = (idx - row * 114) * 8;
      *(v8h*)(vl + row * VLS + c) = *(const v8h*)(vg + row * NPAD + c);
    }
  }
  __syncthreads();

  const _Float16* qb  = q16 + (size_t)bh * NSEQ * HD;
  const float*    pbp = prior + (size_t)bh * NSEQ * NSEQ;

  const int qtend = (part * 15 + 15 < 57) ? part * 15 + 15 : 57;
  for (int qtile = part * 15 + wave; qtile < qtend; qtile += 8) {
    const int qq = qtile * 16 + ln;        // this lane's query column
    const bool qvalid = qq < NSEQ;
    const int qcl = qvalid ? qq : 0;       // clamped: invalid lanes never stored
    // Q^T B-fragment (d x q): elem j -> d = (j<8 ? j : j+8) + 8h (two 16B loads)
    const v16h qfrag = __builtin_shufflevector(*(const v8h*)(qb + (size_t)qcl * HD + h * 8), *(const v8h*)(qb + (size_t)qcl * HD + h * 8 + 16), 0, 1, 2, 3, 4, 5, 6, 7, 8, 9, 10, 11, 12, 13, 14, 15);
    const float* prior_row = pbp + (size_t)qcl * NSEQ;

    v8f Oa = {}, Ob = {};                  // O^T d-tiles [0..15], [16..31]
    float m_run = -3.0e38f;
    float l_run = 0.0f;

    for (int kc = 0; kc < 896; kc += 32) {
      __builtin_prefetch(prior_row + kc + 32, 0, 1);   // next chunk of prior stream
      attn_chunk<true>(kc, h, ln, kl, vl, prior_row, qfrag, Oa, Ob, m_run, l_run);
    }
    attn_chunk<false>(896, h, ln, kl, vl, prior_row, qfrag, Oa, Ob, m_run, l_run);

    const float inv = 1.0f / l_run;
    if (qvalid) {
      // ctx is [B, N, C] f16; lane stores d = 8h..8h+7 (Oa) and +16 (Ob)
      _Float16* crow = ctx + (size_t)(b * NSEQ + qq) * CH + hh * HD + 8 * h;
      v8h oa, ob;
      #pragma unroll
      for (int r = 0; r < 8; ++r) {
        oa[r] = (_Float16)(Oa[r] * inv);
        ob[r] = (_Float16)(Ob[r] * inv);
      }
      *(v8h*)(crow)      = oa;
      *(v8h*)(crow + 16) = ob;
    }
  }
}

// ---------------------------------------------------------------------------
// Kernel 3: output projection  out = ctx @ Wo^T + bo   (fp32 result)
// ---------------------------------------------------------------------------
__global__ __launch_bounds__(128) void out_proj_kernel(
    const _Float16* __restrict__ ctx, const float* __restrict__ Wo,
    const float* __restrict__ bo, float* __restrict__ out)
{
  const int lane = threadIdx.x;
  const int h  = lane >> 4;
  const int ln = lane & 15;
  const int ntile = blockIdx.x * 4 + threadIdx.y;  // 0..15
  const int mtile = blockIdx.y;                    // 0..899
  const int rowbase = mtile * 16;
  const int colbase = ntile * 16;

  v8f acc = {};
  const _Float16* arowb = ctx + (size_t)(rowbase + ln) * CH;
  const float*    browb = Wo  + (size_t)(colbase + ln) * CH;
  for (int kc = 0; kc < CH; kc += 32) {
    v16h a, bfr;
    v8h alo = *(const v8h*)(arowb + kc + h * 8);
    v8h ahi = *(const v8h*)(arowb + kc + h * 8 + 16);
    #pragma unroll
    for (int j = 0; j < 8; ++j) { a[j] = alo[j]; a[j + 8] = ahi[j]; }
    v4f b0 = *(const v4f*)(browb + kc + h * 8);
    v4f b1 = *(const v4f*)(browb + kc + h * 8 + 4);
    v4f b2 = *(const v4f*)(browb + kc + h * 8 + 16);
    v4f b3 = *(const v4f*)(browb + kc + h * 8 + 20);
    #pragma unroll
    for (int j = 0; j < 4; ++j) {
      bfr[j]      = (_Float16)b0[j];
      bfr[j + 4]  = (_Float16)b1[j];
      bfr[j + 8]  = (_Float16)b2[j];
      bfr[j + 12] = (_Float16)b3[j];
    }
    acc = __builtin_amdgcn_wmma_f32_16x16x32_f16(false, a, false, bfr, (short)0, acc, false, false);
  }
  const float bval = bo[colbase + ln];
  #pragma unroll
  for (int r = 0; r < 8; ++r) {
    int row = rowbase + r + 8 * h;   // row = b*900 + n, matches [B,N,C] output
    out[(size_t)row * CH + colbase + ln] = acc[r] + bval;
  }
}

// ---------------------------------------------------------------------------
extern "C" void kernel_launch(void* const* d_in, const int* in_sizes, int n_in,
                              void* d_out, int out_size, void* d_ws, size_t ws_size,
                              hipStream_t stream) {
  (void)in_sizes; (void)n_in; (void)out_size; (void)ws_size;
  const float* qin   = (const float*)d_in[0];
  const float* kin   = (const float*)d_in[1];
  const float* vin   = (const float*)d_in[2];
  const float* prior = (const float*)d_in[3];
  const float* Wq = (const float*)d_in[4];
  const float* bq = (const float*)d_in[5];
  const float* Wk = (const float*)d_in[6];
  const float* bk = (const float*)d_in[7];
  const float* Wv = (const float*)d_in[8];
  const float* bv = (const float*)d_in[9];
  const float* Wo = (const float*)d_in[10];
  const float* bo = (const float*)d_in[11];

  char* ws = (char*)d_ws;
  const size_t qk_bytes = (size_t)NB * NH * NSEQ * HD * sizeof(_Float16);  // 7.37 MB
  const size_t vT_bytes = (size_t)NB * NH * HD * NPAD * sizeof(_Float16);  // 7.47 MB
  _Float16* q16  = (_Float16*)(ws);
  _Float16* k16  = (_Float16*)(ws + qk_bytes);
  _Float16* vT16 = (_Float16*)(ws + 2 * qk_bytes);
  _Float16* ctx  = (_Float16*)(ws + 2 * qk_bytes + vT_bytes);              // 7.37 MB

  dim3 pblk(32, 4), pgrd(4, NSEQ, 3);
  qkv_proj_kernel<<<pgrd, pblk, 0, stream>>>(qin, kin, vin, Wq, bq, Wk, bk, Wv, bv,
                                             q16, k16, vT16);

  dim3 ablk(32, 8), agrd(4, NB * NH);           // 4 query-tile ranges x 128 (b,h)
  const size_t lds_bytes = (size_t)LDS_HALVES * sizeof(_Float16);  // ~128 KB
  attn_kernel<<<agrd, ablk, lds_bytes, stream>>>(q16, k16, vT16, prior, ctx);

  dim3 oblk(32, 4), ogrd(4, NSEQ);
  out_proj_kernel<<<ogrd, oblk, 0, stream>>>(ctx, Wo, bo, (float*)d_out);
}
